// GCN_basic_35871566856587
// MI455X (gfx1250) — matrix operand order repaired and checked by reading, not compiled, not run on hardware
//
#include <hip/hip_runtime.h>
#include <hip/hip_bf16.h>

// ---------------------------------------------------------------------------
// GCN: h1 = relu(spmm(x@W1)+b1); h2 = relu(spmm(h1@W2)+b2);
//      h3 = relu(spmm(h2@W3)+b3); out = h3@Wd + bd
// GEMMs: bf16 WMMA (v_wmma_f32_16x16x32_bf16), fp32 accumulate. Weights
//        pre-transposed to bf16; activations stored bf16 between stages
//        (every value passes through bf16 at the WMMA anyway), halving the
//        dominant SpMM gather traffic and making A/B staging pure b128 copies.
// SpMM: on-device COO->CSR build each launch (graph-capture safe), then
//       uint2 (4xbf16) row-gather; feature matrix is L2-resident (51MB).
// ---------------------------------------------------------------------------

#define GCN_N      100000
#define GCN_E      1600000
#define GCN_NFEAT  512
#define GCN_H2     256
#define GCN_H3     256
#define GCN_H4     128
#define GCN_NCLASS 16

typedef __attribute__((ext_vector_type(8)))  float  v8f;
typedef __attribute__((ext_vector_type(16))) __bf16 v16bf;

__device__ __forceinline__ unsigned short f32_to_bf16(float f) {
    unsigned int u = __float_as_uint(f);
    u += 0x7FFFu + ((u >> 16) & 1u);      // round-to-nearest-even
    return (unsigned short)(u >> 16);
}
__device__ __forceinline__ unsigned int pack_bf16x2(float lo, float hi) {
    unsigned int a = __float_as_uint(lo);
    unsigned int b = __float_as_uint(hi);
    a += 0x7FFFu + ((a >> 16) & 1u);
    b += 0x7FFFu + ((b >> 16) & 1u);
    return (a >> 16) | (b & 0xFFFF0000u);
}

// ---------------------------------------------------------------------------
// Weight prep: W[K x Nc] f32 row-major -> WT[Nc x K] bf16 (transposed).
// Total weights < 1MB, runs once per launch.
// ---------------------------------------------------------------------------
__global__ void convert_WT(const float* __restrict__ W, unsigned short* __restrict__ WT,
                           int K, int Nc) {
    int idx = blockIdx.x * 256 + threadIdx.x;
    if (idx < K * Nc) {
        int k = idx / Nc, n = idx % Nc;          // coalesced read of W
        WT[(long)n * K + k] = f32_to_bf16(W[idx]);
    }
}

// ---------------------------------------------------------------------------
// bf16 WMMA GEMM: C[M x Nc] = A[M x K] * BT^T (+ bias), fp32 accumulate.
// A: f32 or bf16 row-major (A_BF16). BT: bf16 [Nc x K] (pre-transposed).
// C: f32 or bf16 (OUT_BF16).
// Block: 256 threads (8 wave32), tile = 128 rows x (TN*16) cols.
// ---------------------------------------------------------------------------
#define APITCH 40   // LDS row pitch in halfs (80B rows: 16B-aligned frags, bank stride 20)

template <int TN, bool A_BF16, bool OUT_BF16, bool HASBIAS>
__global__ __launch_bounds__(256) void gemm_bf16_wmma(
    const void* __restrict__ Av, const unsigned short* __restrict__ BT,
    void* __restrict__ Cv, int M, int K, int Nc,
    const float* __restrict__ bias)
{
    constexpr int TILEN = TN * 16;
    __shared__ unsigned short As[128 * APITCH];     // A tile: 128 x 32 (row-major bf16)
    __shared__ unsigned short Bs[TILEN * APITCH];   // B tile transposed: TILEN x 32

    const int tid  = threadIdx.x;
    const int lane = tid & 31;
    const int wv   = tid >> 5;
    const int m0   = blockIdx.x * 128;
    const int n0   = blockIdx.y * TILEN;

    v8f acc[TN];
    #pragma unroll
    for (int j = 0; j < TN; ++j) acc[j] = v8f{0.f,0.f,0.f,0.f,0.f,0.f,0.f,0.f};

    const int mlocal = wv * 16 + (lane & 15);
    const int kbaseA = (lane < 16) ? 0 : 8;    // halfs
    const int kbaseB = (lane < 16) ? 0 : 8;    // halfs

    for (int kt = 0; kt < K; kt += 32) {
        // --- stage A tile (128x32 bf16) ---
        if constexpr (A_BF16) {
            const unsigned short* Ab = (const unsigned short*)Av;
            #pragma unroll
            for (int i = 0; i < 2; ++i) {            // 512 uint4 copies
                int idx = tid + i * 256;
                int r = idx >> 2, q = (idx & 3) << 3;
                int row = m0 + r; if (row >= M) row = M - 1;
                uint4 v = *reinterpret_cast<const uint4*>(&Ab[(long)row * K + kt + q]);
                *reinterpret_cast<uint4*>(&As[r * APITCH + q]) = v;
            }
        } else {
            const float* Af = (const float*)Av;
            #pragma unroll
            for (int i = 0; i < 8; ++i) {            // 2048 bf16-pairs
                int p = tid + i * 256;
                int r = p >> 4, c = (p & 15) << 1;
                int row = m0 + r; if (row >= M) row = M - 1;
                const float2 f = *reinterpret_cast<const float2*>(&Af[(long)row * K + kt + c]);
                *reinterpret_cast<unsigned int*>(&As[r * APITCH + c]) = pack_bf16x2(f.x, f.y);
            }
        }
        // --- stage B tile: direct bf16 b128 copies from pre-transposed WT ---
        #pragma unroll 1
        for (int idx = tid; idx < TILEN * 4; idx += 256) {   // uint4 = 8 halfs
            int n = idx >> 2, q = (idx & 3) << 3;
            uint4 v = *reinterpret_cast<const uint4*>(&BT[(long)(n0 + n) * K + kt + q]);
            *reinterpret_cast<uint4*>(&Bs[n * APITCH + q]) = v;
        }
        __syncthreads();

        // --- A fragment (16x32): lanes 0-15 row m K0-7/16-23; lanes 16-31 K8-15/24-31 ---
        union { uint4 q[2]; v16bf bf; } fa;
        {
            const uint4* pa = reinterpret_cast<const uint4*>(&As[mlocal * APITCH + kbaseA]);
            fa.q[0] = pa[0];   // K = kbaseA .. kbaseA+7
            fa.q[1] = pa[2];   // K = kbaseA+16 .. kbaseA+23
        }
        #pragma unroll
        for (int j = 0; j < TN; ++j) {
            // --- B fragment (32x16): lane column n, K kbaseB..+7 / kbaseB+16..+23 ---
            union { uint4 q[2]; v16bf bf; } fb;
            const int nlocal = j * 16 + (lane & 15);
            const uint4* pb = reinterpret_cast<const uint4*>(&Bs[nlocal * APITCH + kbaseB]);
            fb.q[0] = pb[0];
            fb.q[1] = pb[2];
            acc[j] = __builtin_amdgcn_wmma_f32_16x16x32_bf16(
                false, fa.bf, false, fb.bf, (short)0, acc[j], false, false);
        }
        __syncthreads();
    }

    // --- epilogue: C/D layout (8 VGPRs, M = i + 8*(lane>=16), N = lane&15) ---
    const int rbase = m0 + wv * 16 + ((lane >> 4) * 8);
    #pragma unroll
    for (int j = 0; j < TN; ++j) {
        const int coln = n0 + j * 16 + (lane & 15);
        const float bv = HASBIAS ? bias[coln] : 0.f;
        #pragma unroll
        for (int i = 0; i < 8; ++i) {
            int r = rbase + i;
            if (r < M) {
                float o = acc[j][i] + bv;
                if constexpr (OUT_BF16)
                    ((unsigned short*)Cv)[(long)r * Nc + coln] = f32_to_bf16(o);
                else
                    ((float*)Cv)[(long)r * Nc + coln] = o;
            }
        }
    }
}

// ---------------------------------------------------------------------------
// CSR build: histogram -> block scan -> one-block scan of partials -> scatter
// ---------------------------------------------------------------------------
__global__ void zero_ints(int* p, int n) {
    int i = blockIdx.x * blockDim.x + threadIdx.x;
    if (i < n) p[i] = 0;
}
__global__ void count_rows(const int* __restrict__ rows, int* counts, int e) {
    int i = blockIdx.x * blockDim.x + threadIdx.x;
    if (i < e) atomicAdd(&counts[rows[i]], 1);
}
__global__ void scan256(const int* __restrict__ in, int* __restrict__ out,
                        int* __restrict__ sums, int n) {
    __shared__ int tmp[256];
    int i = blockIdx.x * 256 + threadIdx.x;
    tmp[threadIdx.x] = (i < n) ? in[i] : 0;
    __syncthreads();
    #pragma unroll
    for (int off = 1; off < 256; off <<= 1) {
        int t = (threadIdx.x >= off) ? tmp[threadIdx.x - off] : 0;
        __syncthreads();
        tmp[threadIdx.x] += t;
        __syncthreads();
    }
    if (i < n) out[i] = tmp[threadIdx.x];                 // inclusive, block-local
    if (threadIdx.x == 255) sums[blockIdx.x] = tmp[255];  // block total
}
__global__ __launch_bounds__(512) void scan_sums(int* sums, int nb) {  // nb <= 512
    __shared__ int tmp[512];
    int t = threadIdx.x;
    int v = (t < nb) ? sums[t] : 0;
    tmp[t] = v;
    __syncthreads();
    #pragma unroll
    for (int off = 1; off < 512; off <<= 1) {
        int x = (t >= off) ? tmp[t - off] : 0;
        __syncthreads();
        tmp[t] += x;
        __syncthreads();
    }
    if (t < nb) sums[t] = tmp[t] - v;   // exclusive
}
__global__ void finalize_rptr(const int* __restrict__ incl, const int* __restrict__ sums,
                              int* __restrict__ rptr, int n) {
    int i = blockIdx.x * 256 + threadIdx.x;
    if (i < n) rptr[i + 1] = incl[i] + sums[i >> 8];
    if (i == 0) rptr[0] = 0;
}
__global__ void copy_ints(const int* __restrict__ a, int* __restrict__ b, int n) {
    int i = blockIdx.x * blockDim.x + threadIdx.x;
    if (i < n) b[i] = a[i];
}
__global__ void scatter_edges(const int* __restrict__ rows, const int* __restrict__ cols,
                              const float* __restrict__ vals, int* cursor,
                              int* __restrict__ cols_s, float* __restrict__ vals_s, int e) {
    int i = blockIdx.x * blockDim.x + threadIdx.x;
    if (i < e) {
        int r = rows[i];
        int p = atomicAdd(&cursor[r], 1);
        cols_s[p] = cols[i];
        vals_s[p] = vals[i];
    }
}

// ---------------------------------------------------------------------------
// CSR SpMM + bias + relu on bf16 activations, fp32 accumulate.
// F/4 threads per row (uint2 = 4 x bf16 per gather), 256-thread blocks.
// ---------------------------------------------------------------------------
template <int F>
__global__ __launch_bounds__(256) void spmm_bias_relu_bf16(
    const int* __restrict__ rptr, const int* __restrict__ cols,
    const float* __restrict__ vals, const unsigned short* __restrict__ Hin,
    const float* __restrict__ bias, unsigned short* __restrict__ Hout, int nrows)
{
    constexpr int TPR = F / 4;        // threads per row
    constexpr int RPB = 256 / TPR;    // rows per block
    const int row = blockIdx.x * RPB + threadIdx.x / TPR;
    const int f   = (threadIdx.x % TPR) * 4;
    if (row >= nrows) return;
    const int s = rptr[row], e = rptr[row + 1];
    float a0 = 0.f, a1 = 0.f, a2 = 0.f, a3 = 0.f;
    for (int i = s; i < e; ++i) {
        const int   c = cols[i];
        const float v = vals[i];
        const uint2 u = *reinterpret_cast<const uint2*>(&Hin[(long)c * F + f]);
        a0 = fmaf(v, __uint_as_float(u.x << 16), a0);
        a1 = fmaf(v, __uint_as_float(u.x & 0xFFFF0000u), a1);
        a2 = fmaf(v, __uint_as_float(u.y << 16), a2);
        a3 = fmaf(v, __uint_as_float(u.y & 0xFFFF0000u), a3);
    }
    const float4 b = *reinterpret_cast<const float4*>(&bias[f]);
    a0 = fmaxf(a0 + b.x, 0.f);
    a1 = fmaxf(a1 + b.y, 0.f);
    a2 = fmaxf(a2 + b.z, 0.f);
    a3 = fmaxf(a3 + b.w, 0.f);
    uint2 o;
    o.x = pack_bf16x2(a0, a1);
    o.y = pack_bf16x2(a2, a3);
    *reinterpret_cast<uint2*>(&Hout[(long)row * F + f]) = o;
}

// ---------------------------------------------------------------------------
extern "C" void kernel_launch(void* const* d_in, const int* in_sizes, int n_in,
                              void* d_out, int out_size, void* d_ws, size_t ws_size,
                              hipStream_t stream) {
    (void)in_sizes; (void)n_in; (void)out_size; (void)ws_size;
    const float* x        = (const float*)d_in[0];
    const int*   edge_row = (const int*)  d_in[1];
    const int*   edge_col = (const int*)  d_in[2];
    const float* edge_val = (const float*)d_in[3];
    const float* W1 = (const float*)d_in[4];
    const float* b1 = (const float*)d_in[5];
    const float* W2 = (const float*)d_in[6];
    const float* b2 = (const float*)d_in[7];
    const float* W3 = (const float*)d_in[8];
    const float* b3 = (const float*)d_in[9];
    const float* Wd = (const float*)d_in[10];
    const float* bd = (const float*)d_in[11];
    float* out = (float*)d_out;

    const int N = GCN_N, E = GCN_E;

    // ---- carve workspace ----
    char* ws = (char*)d_ws;
    size_t off = 0;
    auto carve = [&](size_t bytes) -> void* {
        void* p = ws + off;
        off = (off + bytes + 255) & ~(size_t)255;
        return p;
    };
    unsigned short* tbuf   = (unsigned short*)carve((size_t)N * 256 * 2);  // GEMM out (bf16)
    unsigned short* hbuf   = (unsigned short*)carve((size_t)N * 256 * 2);  // SpMM out (bf16)
    int*            rptr   = (int*)           carve((size_t)(N + 1) * sizeof(int));
    int*            cursor = (int*)           carve((size_t)N * sizeof(int));
    int*            incl   = (int*)           carve((size_t)N * sizeof(int));
    int*            sums   = (int*)           carve(1024 * sizeof(int));
    int*            cols_s = (int*)           carve((size_t)E * sizeof(int));
    float*          vals_s = (float*)         carve((size_t)E * sizeof(float));
    unsigned short* wt1    = (unsigned short*)carve((size_t)GCN_NFEAT * GCN_H2 * 2);
    unsigned short* wt2    = (unsigned short*)carve((size_t)GCN_H2 * GCN_H3 * 2);
    unsigned short* wt3    = (unsigned short*)carve((size_t)GCN_H3 * GCN_H4 * 2);
    unsigned short* wt4    = (unsigned short*)carve((size_t)GCN_H4 * GCN_NCLASS * 2);

    const int NB = (N + 255) / 256;  // 391 scan blocks

    // ---- weight prep (bf16 transposed) ----
    convert_WT<<<(GCN_NFEAT * GCN_H2 + 255) / 256, 256, 0, stream>>>(W1, wt1, GCN_NFEAT, GCN_H2);
    convert_WT<<<(GCN_H2 * GCN_H3 + 255) / 256, 256, 0, stream>>>(W2, wt2, GCN_H2, GCN_H3);
    convert_WT<<<(GCN_H3 * GCN_H4 + 255) / 256, 256, 0, stream>>>(W3, wt3, GCN_H3, GCN_H4);
    convert_WT<<<(GCN_H4 * GCN_NCLASS + 255) / 256, 256, 0, stream>>>(Wd, wt4, GCN_H4, GCN_NCLASS);

    // ---- COO -> CSR (rebuilt every launch: no cached state) ----
    zero_ints<<<NB, 256, 0, stream>>>(cursor, N);  // cursor doubles as counts
    count_rows<<<(E + 255) / 256, 256, 0, stream>>>(edge_row, cursor, E);
    scan256<<<NB, 256, 0, stream>>>(cursor, incl, sums, N);
    scan_sums<<<1, 512, 0, stream>>>(sums, NB);
    finalize_rptr<<<NB, 256, 0, stream>>>(incl, sums, rptr, N);
    copy_ints<<<NB, 256, 0, stream>>>(rptr, cursor, N);
    scatter_edges<<<(E + 255) / 256, 256, 0, stream>>>(edge_row, edge_col, edge_val,
                                                       cursor, cols_s, vals_s, E);

    const int MT = (N + 127) / 128;  // 782 row tiles

    // ---- layer 1: t = x @ W1 (K=512, Nc=256, 256-wide tile: x read once) ----
    gemm_bf16_wmma<16, false, true, false><<<dim3(MT, 1), 256, 0, stream>>>(
        x, wt1, tbuf, N, GCN_NFEAT, GCN_H2, nullptr);
    spmm_bias_relu_bf16<256><<<(N + 3) / 4, 256, 0, stream>>>(
        rptr, cols_s, vals_s, tbuf, b1, hbuf, N);

    // ---- layer 2: t = h1 @ W2 (K=256, Nc=256) ----
    gemm_bf16_wmma<16, true, true, false><<<dim3(MT, 1), 256, 0, stream>>>(
        hbuf, wt2, tbuf, N, GCN_H2, GCN_H3, nullptr);
    spmm_bias_relu_bf16<256><<<(N + 3) / 4, 256, 0, stream>>>(
        rptr, cols_s, vals_s, tbuf, b2, hbuf, N);

    // ---- layer 3: t = h2 @ W3 (K=256, Nc=128) ----
    gemm_bf16_wmma<8, true, true, false><<<dim3(MT, 1), 256, 0, stream>>>(
        hbuf, wt3, tbuf, N, GCN_H3, GCN_H4, nullptr);
    spmm_bias_relu_bf16<128><<<(N + 7) / 8, 256, 0, stream>>>(
        rptr, cols_s, vals_s, tbuf, b3, hbuf, N);

    // ---- output: out = h3 @ Wd + bd (K=128, Nc=16, f32 out) ----
    gemm_bf16_wmma<1, true, false, true><<<dim3(MT, 1), 256, 0, stream>>>(
        hbuf, wt4, out, N, GCN_H4, GCN_NCLASS, bd);
}
